// EncodeProcessDecode_57028575756313
// MI455X (gfx1250) — matrix operand order repaired and checked by reading, not compiled, not run on hardware
//
#include <hip/hip_runtime.h>

// ---------------------------------------------------------------------------
// MeshGraphNets-style EncodeProcessDecode for MI455X (gfx1250, wave32).
// GEMMs: bf16 WMMA 16x16x32, f32 accumulate. Row gathers: TDM gather-mode
// (tensor_load_to_lds) with LDS pad-interleave directly into WMMA A-tiles.
// ---------------------------------------------------------------------------

#define NN   50000
#define NE   400000
#define LATD 128
#define EPSLN 1e-5f

typedef __bf16 bf16;
typedef bf16  v16bf __attribute__((ext_vector_type(16)));
typedef bf16  v8bf  __attribute__((ext_vector_type(8)));
typedef float v8f   __attribute__((ext_vector_type(8)));
typedef unsigned int u32x4 __attribute__((ext_vector_type(4)));
typedef int   i32x8 __attribute__((ext_vector_type(8)));
typedef int   i32x4 __attribute__((ext_vector_type(4)));

union ABf { v16bf v; v8bf h[2]; };

// ---- WMMA fragment loaders (layouts per CDNA5 ISA 7.12.2) ------------------

// A fragment 16x32 bf16 from LDS tile [16][strideK]: two b128 LDS reads/lane.
__device__ inline v16bf load_a_frag(const bf16* sA, int strideK, int kt, int lane) {
    int m    = lane & 15;
    int half = lane >> 4;
    const bf16* p = sA + m * strideK + kt * 32 + half * 8;
    ABf u;
    u.h[0] = *(const v8bf*)(p);
    u.h[1] = *(const v8bf*)(p + 16);
    return u.v;
}

// B fragment 32x16 bf16 from fragment-packed weights: one 32B load per lane.
__device__ inline v16bf load_b_frag(const bf16* Wp, int kt, int nt, int lane) {
    return *(const v16bf*)(Wp + ((((kt << 3) + nt) * 32 + lane) << 4));
}

__device__ inline v8f wmma_bf16(v16bf a, v16bf b, v8f c) {
    return __builtin_amdgcn_wmma_f32_16x16x32_bf16(
        false, a, false, b, (short)0, c, false, false);
}

__device__ inline void store_d_bf16_relu(v8f acc, const float* bias,
                                         bf16* dst, int n0, int lane) {
    int n  = n0 + (lane & 15);
    int m0 = (lane >> 4) * 8;
    float bv = bias[n];
#pragma unroll
    for (int r = 0; r < 8; ++r) {
        float v = acc[r] + bv;
        dst[(m0 + r) * LATD + n] = (bf16)(v > 0.f ? v : 0.f);
    }
}

__device__ inline void store_d_f32_relu(v8f acc, const float* bias,
                                        float* dst, int n0, int lane) {
    int n  = n0 + (lane & 15);
    int m0 = (lane >> 4) * 8;
    float bv = bias[n];
#pragma unroll
    for (int r = 0; r < 8; ++r) {
        float v = acc[r] + bv;
        dst[(m0 + r) * LATD + n] = v > 0.f ? v : 0.f;
    }
}

// ---- TDM: async tensor load into LDS (CDNA5 ISA ch.8) ----------------------
__device__ inline void tdm_issue(u32x4 g0, i32x8 g1, i32x4 g2, i32x4 g3) {
#if defined(__clang_major__) && (__clang_major__ >= 23)
    i32x8 g45 = {0, 0, 0, 0, 0, 0, 0, 0};
    __builtin_amdgcn_tensor_load_to_lds(g0, g1, g2, g3, g45, 0);
#else
    __builtin_amdgcn_tensor_load_to_lds(g0, g1, g2, g3, 0);
#endif
}

// Load a 16x128 bf16 tile (gather-mode rows via g2/g3 indices, or contiguous
// 2D tile) into LDS. Optional LDS padding interleaves 256B rows into an
// A-tile of larger row stride: pad_interval = 64 DWORDs (one row), then skip
// padAmtCode+1 DWORDs.
__device__ inline void tdm_load16x128(const void* base, unsigned ldsAddr,
                                      unsigned dim1Rows, int gatherMode,
                                      int padEnable, unsigned padAmtCode,
                                      i32x4 g2, i32x4 g3) {
    unsigned long long ga = (unsigned long long)base;
    u32x4 g0;
    g0[0] = 1u | ((unsigned)gatherMode << 31);      // count=1, 16-bit gather idx
    g0[1] = ldsAddr;                                // LDS byte address
    g0[2] = (unsigned)ga;                           // global_addr[31:0]
    g0[3] = (unsigned)(ga >> 32) | (2u << 30);      // global_addr[56:32], type=2
    i32x8 g1;
    unsigned w0 = (1u << 16);                       // data_size = 2 bytes
    if (padEnable) w0 |= (1u << 20) | (5u << 22) | (padAmtCode << 25);
    g1[0] = (int)w0;
    g1[1] = (int)(128u << 16);                      // tensor_dim0 = 128
    g1[2] = (int)(dim1Rows << 16);                  // tensor_dim1[15:0]
    g1[3] = (int)((dim1Rows >> 16) | (128u << 16)); // dim1[31:16] | tile_dim0=128
    g1[4] = 16;                                     // tile_dim1 (= #gather rows)
    g1[5] = 128;                                    // tensor_dim0_stride
    g1[6] = 0;
    g1[7] = 0;
    tdm_issue(g0, g1, g2, g3);
}

// ---- LayerNorm over 128 features, wave32 shuffle reduction, 2 rows/wave ----
__device__ inline void layernorm_rows(float* sH, const float* g,
                                      const float* be, int wave, int lane) {
#pragma unroll
    for (int rr = 0; rr < 2; ++rr) {
        int row = wave * 2 + rr;
        float* hp = sH + row * LATD;
        float4 x = *(const float4*)(hp + lane * 4);
        float s = x.x + x.y + x.z + x.w;
#pragma unroll
        for (int off = 16; off > 0; off >>= 1) s += __shfl_xor(s, off, 32);
        float mu = s * (1.0f / LATD);
        float d0 = x.x - mu, d1 = x.y - mu, d2 = x.z - mu, d3 = x.w - mu;
        float v = d0 * d0 + d1 * d1 + d2 * d2 + d3 * d3;
#pragma unroll
        for (int off = 16; off > 0; off >>= 1) v += __shfl_xor(v, off, 32);
        float rs = rsqrtf(v * (1.0f / LATD) + EPSLN);
        int n = lane * 4;
        float4 gg = *(const float4*)(g + n);
        float4 bb = *(const float4*)(be + n);
        hp[n + 0] = d0 * rs * gg.x + bb.x;
        hp[n + 1] = d1 * rs * gg.y + bb.y;
        hp[n + 2] = d2 * rs * gg.z + bb.z;
        hp[n + 3] = d3 * rs * gg.w + bb.w;
    }
}

// ---- Fused Linear->ReLU->Linear->ReLU->LayerNorm on a 16-row tile ----------
template <int K1>
__device__ inline void mlp2_ln(const bf16* sA, bf16* sH1, float* sH2,
                               const bf16* w1p, const float* b1,
                               const bf16* w2p, const float* b2,
                               const float* g, const float* be) {
    const int lane = threadIdx.x & 31;
    const int wave = threadIdx.x >> 5;
    v8f acc = {};
#pragma unroll
    for (int kt = 0; kt < K1 / 32; ++kt)
        acc = wmma_bf16(load_a_frag(sA, K1, kt, lane),
                        load_b_frag(w1p, kt, wave, lane), acc);
    store_d_bf16_relu(acc, b1, sH1, wave * 16, lane);
    __syncthreads();
    v8f acc2 = {};
#pragma unroll
    for (int kt = 0; kt < 4; ++kt)
        acc2 = wmma_bf16(load_a_frag(sH1, LATD, kt, lane),
                         load_b_frag(w2p, kt, wave, lane), acc2);
    store_d_f32_relu(acc2, b2, sH2, wave * 16, lane);
    __syncthreads();
    layernorm_rows(sH2, g, be, wave, lane);
    __syncthreads();
}

// Stage 8 f32 -> 8 bf16 into LDS (two coalesced float4 global loads).
__device__ inline void stage8(const float* s, bf16* d) {
    float4 u = ((const float4*)s)[0];
    float4 v = ((const float4*)s)[1];
    d[0] = (bf16)u.x; d[1] = (bf16)u.y; d[2] = (bf16)u.z; d[3] = (bf16)u.w;
    d[4] = (bf16)v.x; d[5] = (bf16)v.y; d[6] = (bf16)v.z; d[7] = (bf16)v.w;
}

// ---- Weight packing: f32 row-major [K][128] -> bf16 B-fragment order -------
__global__ __launch_bounds__(256) void pack_w_kernel(const float* __restrict__ W,
                                                     bf16* __restrict__ P, int K) {
    int p = blockIdx.x * 256 + threadIdx.x;
    if (p >= K * LATD) return;
    int i    = p & 15;
    int lane = (p >> 4) & 31;
    int fr   = p >> 9;
    int nt   = fr & 7;
    int kt   = fr >> 3;
    int n = (nt << 4) | (lane & 15);
    int k = kt * 32 + ((i >> 3) << 4) + ((lane >> 4) << 3) + (i & 7);
    P[p] = (bf16)W[k * LATD + n];
}

__global__ __launch_bounds__(256) void zero_kernel(float* p, int n) {
    int i = blockIdx.x * 256 + threadIdx.x;
    if (i < n) p[i] = 0.f;
}

// ---- Encoder: tiny first layer (DIN=3/4) in VALU, second layer in WMMA -----
template <int DIN>
__global__ __launch_bounds__(256) void encode_kernel(
    const float* __restrict__ xin, const float* __restrict__ w1,
    const float* __restrict__ b1, const bf16* __restrict__ w2p,
    const float* __restrict__ b2, const float* __restrict__ g,
    const float* __restrict__ be, float* __restrict__ hout,
    bf16* __restrict__ houtbf) {
    __shared__ __align__(16) bf16  sH1[16 * LATD];
    __shared__ __align__(16) float sH2[16 * LATD];
    const int r0 = blockIdx.x * 16;
    const int t = threadIdx.x;
    {
        int row = t >> 4, c0 = (t & 15) * 8;
        const float* xr = xin + (size_t)(r0 + row) * DIN;
        float xv[DIN];
#pragma unroll
        for (int k = 0; k < DIN; ++k) xv[k] = xr[k];
#pragma unroll
        for (int j = 0; j < 8; ++j) {
            int n = c0 + j;
            float v = b1[n];
#pragma unroll
            for (int k = 0; k < DIN; ++k) v += xv[k] * w1[k * LATD + n];
            sH1[row * LATD + n] = (bf16)(v > 0.f ? v : 0.f);
        }
    }
    __syncthreads();
    const int lane = t & 31, wave = t >> 5;
    v8f acc = {};
#pragma unroll
    for (int kt = 0; kt < 4; ++kt)
        acc = wmma_bf16(load_a_frag(sH1, LATD, kt, lane),
                        load_b_frag(w2p, kt, wave, lane), acc);
    store_d_f32_relu(acc, b2, sH2, wave * 16, lane);
    __syncthreads();
    layernorm_rows(sH2, g, be, wave, lane);
    __syncthreads();
    for (int i = t; i < 16 * LATD; i += 256) {
        float v = sH2[i];
        hout[(size_t)r0 * LATD + i]   = v;
        houtbf[(size_t)r0 * LATD + i] = (bf16)v;
    }
}

// ---- Edge MLP: MODE 0 = message (cat[h[dst],h[src],e]) -> atomic agg[dst]
//                MODE 1 = edge update (cat[h[src],h[dst],e]) -> e += LN-MLP --
// A-tile staged entirely by three TDM descriptors (2 gathers + 1 tile),
// pad-interleaved into the 768B-stride [16][384] LDS tile.
template <int MODE>
__global__ __launch_bounds__(256) void edge_mlp_kernel(
    const bf16* __restrict__ hbf, const bf16* __restrict__ ebf,
    const float* __restrict__ e,
    const int* __restrict__ src, const int* __restrict__ dst,
    const bf16* __restrict__ w1p, const float* __restrict__ b1,
    const bf16* __restrict__ w2p, const float* __restrict__ b2,
    const float* __restrict__ g, const float* __restrict__ be,
    float* __restrict__ outp, bf16* __restrict__ ebf_out) {
    __shared__ __align__(16) bf16  sA[16 * 384];
    __shared__ __align__(16) bf16  sH1[16 * LATD];
    __shared__ __align__(16) float sH2[16 * LATD];
    __shared__ int sIdx[16];
    const int e0 = blockIdx.x * 16;
    const int t = threadIdx.x;
    if (MODE == 0 && (t & 15) == 0) sIdx[t >> 4] = dst[e0 + (t >> 4)];

    if (t < 32) {
        const int* ia = (MODE == 0) ? dst : src;   // columns 0..127
        const int* ib = (MODE == 0) ? src : dst;   // columns 128..255
        i32x4 g2a, g3a, g2b, g3b;
#pragma unroll
        for (int j = 0; j < 4; ++j) {  // uniform scalar loads, pack 16-bit idx
            g2a[j] = (ia[e0 + 2 * j] & 0xffff)     | (ia[e0 + 2 * j + 1] << 16);
            g3a[j] = (ia[e0 + 8 + 2 * j] & 0xffff) | (ia[e0 + 8 + 2 * j + 1] << 16);
            g2b[j] = (ib[e0 + 2 * j] & 0xffff)     | (ib[e0 + 2 * j + 1] << 16);
            g3b[j] = (ib[e0 + 8 + 2 * j] & 0xffff) | (ib[e0 + 8 + 2 * j + 1] << 16);
        }
        unsigned ldsA = (unsigned)(unsigned long long)(void*)sA;
        i32x4 zz = {0, 0, 0, 0};
        // 256B row, then skip 512B -> 768B row stride: pad_amount code 127.
        tdm_load16x128(hbf, ldsA + 0,   NN, 1, 1, 127u, g2a, g3a);
        tdm_load16x128(hbf, ldsA + 256, NN, 1, 1, 127u, g2b, g3b);
        tdm_load16x128(ebf + (size_t)e0 * LATD, ldsA + 512, NE, 0, 1, 127u, zz, zz);
        __builtin_amdgcn_s_wait_tensorcnt(0);
    }
    __syncthreads();
    mlp2_ln<384>(sA, sH1, sH2, w1p, b1, w2p, b2, g, be);
    if (MODE == 0) {
        for (int i = t; i < 16 * LATD; i += 256) {
            int r = i >> 7, n = i & 127;
            unsafeAtomicAdd(&outp[(size_t)sIdx[r] * LATD + n], sH2[i]);
        }
    } else {
        for (int i = t; i < 16 * LATD; i += 256) {
            size_t o = (size_t)e0 * LATD + i;
            float nv = e[o] + sH2[i];   // residual, in place (disjoint rows)
            outp[o] = nv;
            ebf_out[o] = (bf16)nv;
        }
    }
}

// ---- Node update: h = LN-MLP(cat[agg, h]) + h (in place) -------------------
__global__ __launch_bounds__(256) void node_update_kernel(
    const float* __restrict__ agg, float* __restrict__ h,
    bf16* __restrict__ hbf,
    const bf16* __restrict__ w1p, const float* __restrict__ b1,
    const bf16* __restrict__ w2p, const float* __restrict__ b2,
    const float* __restrict__ g, const float* __restrict__ be) {
    __shared__ __align__(16) bf16  sA[16 * 256];
    __shared__ __align__(16) bf16  sH1[16 * LATD];
    __shared__ __align__(16) float sH2[16 * LATD];
    const int r0 = blockIdx.x * 16;
    const int t = threadIdx.x;
    if (t < 32) {  // TDM tile: h rows into cols 128..255 (512B row stride)
        unsigned ldsA = (unsigned)(unsigned long long)(void*)sA;
        i32x4 zz = {0, 0, 0, 0};
        tdm_load16x128(hbf + (size_t)r0 * LATD, ldsA + 256, NN, 0, 1, 63u, zz, zz);
        __builtin_amdgcn_s_wait_tensorcnt(0);
    }
    {
        int row = t >> 4, part = t & 15;
        stage8(agg + (size_t)(r0 + row) * LATD + part * 8,
               sA + row * 256 + part * 8);
    }
    __syncthreads();
    mlp2_ln<256>(sA, sH1, sH2, w1p, b1, w2p, b2, g, be);
    for (int i = t; i < 16 * LATD; i += 256) {
        size_t o = (size_t)r0 * LATD + i;
        float nv = h[o] + sH2[i];
        h[o] = nv;
        hbf[o] = (bf16)nv;
    }
}

// ---- Decoder: relu(h@W1+b1) via WMMA, then 128->3 dot in VALU --------------
__global__ __launch_bounds__(256) void decode_kernel(
    const bf16* __restrict__ hbf, const bf16* __restrict__ w1p,
    const float* __restrict__ b1, const float* __restrict__ w2,
    const float* __restrict__ b2, float* __restrict__ out) {
    __shared__ __align__(16) bf16  sH0[16 * LATD];
    __shared__ __align__(16) float sH1f[16 * LATD];
    const int r0 = blockIdx.x * 16;
    const int t = threadIdx.x;
    if (t < 32) {  // contiguous tile, no padding
        unsigned ldsA = (unsigned)(unsigned long long)(void*)sH0;
        i32x4 zz = {0, 0, 0, 0};
        tdm_load16x128(hbf + (size_t)r0 * LATD, ldsA, NN, 0, 0, 0u, zz, zz);
        __builtin_amdgcn_s_wait_tensorcnt(0);
    }
    __syncthreads();
    const int lane = t & 31, wave = t >> 5;
    v8f acc = {};
#pragma unroll
    for (int kt = 0; kt < 4; ++kt)
        acc = wmma_bf16(load_a_frag(sH0, LATD, kt, lane),
                        load_b_frag(w1p, kt, wave, lane), acc);
    store_d_f32_relu(acc, b1, sH1f, wave * 16, lane);
    __syncthreads();
    if (t < 48) {
        int row = t / 3, o = t % 3;
        float s = b2[o];
        for (int n = 0; n < LATD; ++n) s += sH1f[row * LATD + n] * w2[n * 3 + o];
        out[(size_t)(r0 + row) * 3 + o] = s;
    }
}

// ---------------------------------------------------------------------------
extern "C" void kernel_launch(void* const* d_in, const int* in_sizes, int n_in,
                              void* d_out, int out_size, void* d_ws, size_t ws_size,
                              hipStream_t stream) {
    const float* x         = (const float*)d_in[0];
    const float* edge_attr = (const float*)d_in[1];
    const int*   eidx      = (const int*)d_in[2];
    const int*   src = eidx;           // edge_index[0] = source j
    const int*   dst = eidx + NE;      // edge_index[1] = target i
    const float* ne_w1 = (const float*)d_in[3];  const float* ne_b1 = (const float*)d_in[4];
    const float* ne_w2 = (const float*)d_in[5];  const float* ne_b2 = (const float*)d_in[6];
    const float* ne_g  = (const float*)d_in[7];  const float* ne_be = (const float*)d_in[8];
    const float* ee_w1 = (const float*)d_in[9];  const float* ee_b1 = (const float*)d_in[10];
    const float* ee_w2 = (const float*)d_in[11]; const float* ee_b2 = (const float*)d_in[12];
    const float* ee_g  = (const float*)d_in[13]; const float* ee_be = (const float*)d_in[14];
    const float* em_w1 = (const float*)d_in[15]; const float* em_b1 = (const float*)d_in[16];
    const float* em_w2 = (const float*)d_in[17]; const float* em_b2 = (const float*)d_in[18];
    const float* em_g  = (const float*)d_in[19]; const float* em_be = (const float*)d_in[20];
    const float* nm_w1 = (const float*)d_in[21]; const float* nm_b1 = (const float*)d_in[22];
    const float* nm_w2 = (const float*)d_in[23]; const float* nm_b2 = (const float*)d_in[24];
    const float* nm_g  = (const float*)d_in[25]; const float* nm_be = (const float*)d_in[26];
    const float* dec_w1 = (const float*)d_in[27]; const float* dec_b1 = (const float*)d_in[28];
    const float* dec_w2 = (const float*)d_in[29]; const float* dec_b2 = (const float*)d_in[30];

    // Workspace: f32 masters + bf16 gather mirrors + packed bf16 weights.
    char* wp = (char*)d_ws;
    auto carve = [&](size_t bytes) {
        void* p = (void*)wp;
        wp += (bytes + 255) & ~(size_t)255;
        return p;
    };
    float* h    = (float*)carve((size_t)NN * LATD * 4);
    float* ebuf = (float*)carve((size_t)NE * LATD * 4);
    float* agg  = (float*)carve((size_t)NN * LATD * 4);
    bf16*  hbf  = (bf16*)carve((size_t)NN * LATD * 2);
    bf16*  ebf  = (bf16*)carve((size_t)NE * LATD * 2);
    bf16* p_ne2  = (bf16*)carve((size_t)128 * LATD * 2);
    bf16* p_ee2  = (bf16*)carve((size_t)128 * LATD * 2);
    bf16* p_em1  = (bf16*)carve((size_t)384 * LATD * 2);
    bf16* p_em2  = (bf16*)carve((size_t)128 * LATD * 2);
    bf16* p_nm1  = (bf16*)carve((size_t)256 * LATD * 2);
    bf16* p_nm2  = (bf16*)carve((size_t)128 * LATD * 2);
    bf16* p_dec1 = (bf16*)carve((size_t)128 * LATD * 2);

    auto pack = [&](const float* W, bf16* P, int K) {
        pack_w_kernel<<<(K * LATD + 255) / 256, 256, 0, stream>>>(W, P, K);
    };
    pack(ne_w2, p_ne2, 128);
    pack(ee_w2, p_ee2, 128);
    pack(em_w1, p_em1, 384);
    pack(em_w2, p_em2, 128);
    pack(nm_w1, p_nm1, 256);
    pack(nm_w2, p_nm2, 128);
    pack(dec_w1, p_dec1, 128);

    // Encode
    encode_kernel<3><<<NN / 16, 256, 0, stream>>>(x, ne_w1, ne_b1, p_ne2, ne_b2,
                                                  ne_g, ne_be, h, hbf);
    encode_kernel<4><<<NE / 16, 256, 0, stream>>>(edge_attr, ee_w1, ee_b1, p_ee2,
                                                  ee_b2, ee_g, ee_be, ebuf, ebf);

    // Process: 5 shared-weight GraphNet blocks.
    // Edge update runs before the in-place node update so both see old h.
    for (int step = 0; step < 5; ++step) {
        zero_kernel<<<(NN * LATD + 255) / 256, 256, 0, stream>>>(agg, NN * LATD);
        edge_mlp_kernel<0><<<NE / 16, 256, 0, stream>>>(hbf, ebf, ebuf, src, dst,
            p_em1, em_b1, p_em2, em_b2, em_g, em_be, agg, nullptr);
        edge_mlp_kernel<1><<<NE / 16, 256, 0, stream>>>(hbf, ebf, ebuf, src, dst,
            p_em1, em_b1, p_em2, em_b2, em_g, em_be, ebuf, ebf);
        node_update_kernel<<<NN / 16, 256, 0, stream>>>(agg, h, hbf,
            p_nm1, nm_b1, p_nm2, nm_b2, nm_g, nm_be);
    }

    // Decode
    decode_kernel<<<NN / 16, 256, 0, stream>>>(hbf, p_dec1, dec_b1, dec_w2,
                                               dec_b2, (float*)d_out);
}
